// DifferentialWindowAttention_55490977464441
// MI455X (gfx1250) — compile-verified, matrix operand order checked by reading, not run on hardware
//
#include <hip/hip_runtime.h>
#include <hip/hip_bf16.h>
#include <math.h>

#define DIM      1024
#define NHEAD    32
#define HD       32
#define HDL      8
#define LITE     256
#define NW       49           // window area
#define NUMWIN   64
#define BATCH    512          // batch * num_windows
#define MTOK     (BATCH * NW) // 25088
#define N1       3840         // 3*DIM + 3*LITE
#define K1       DIM
#define N2       DIM
#define K2       (DIM + LITE) // 1280

// GEMM tiling
#define BM 128
#define BN 128
#define BK 64
#define LDT 72                // padded LDS leading dim (bf16 elems): 144B rows

typedef __bf16 bf16_t;
typedef __attribute__((ext_vector_type(16))) __bf16 v16bf;
typedef __attribute__((ext_vector_type(8)))  __bf16 v8bf;
typedef __attribute__((ext_vector_type(8)))  float  v8f;

// types matching the async-to-LDS builtin signature (per compiler diagnostic:
// param is 'int __attribute__((vector_size(16)))' pointer in AS1 / AS3)
typedef int v4i_t __attribute__((vector_size(16)));
typedef __attribute__((address_space(1))) v4i_t as1_v4i;
typedef __attribute__((address_space(3))) v4i_t as3_v4i;

#if defined(__has_builtin)
#if __has_builtin(__builtin_amdgcn_global_load_async_to_lds_b128)
#define HAVE_ASYNC_LDS 1
#endif
#endif
#ifndef HAVE_ASYNC_LDS
#define HAVE_ASYNC_LDS 0
#endif

// ---------------------------------------------------------------------------
// WMMA fragment helpers (CDNA5 bf16 16x16x32 layouts, ISA 7.12.2)
// A frag: lane<16 -> row=lane, K {k0..k0+7, k0+16..k0+23}
//         lane>=16 -> row=lane-16, K {k0+8..k0+15, k0+24..k0+31}
// B frag (B = W^T, W row-major along K): identical load from W rows.
// ---------------------------------------------------------------------------
__device__ __forceinline__ v16bf frag_load(const bf16_t* __restrict__ tile,
                                           int ld, int lane) {
  const int half = lane >> 4;
  const bf16_t* p = tile + (size_t)(lane & 15) * ld + half * 8;
  v8bf lo = *(const v8bf*)(p);
  v8bf hi = *(const v8bf*)(p + 16);
  v16bf r;
#pragma unroll
  for (int i = 0; i < 8; ++i) { r[i] = lo[i]; r[i + 8] = hi[i]; }
  return r;
}

__device__ __forceinline__ v8f wmma_bf16(v16bf a, v16bf b, v8f c) {
  return __builtin_amdgcn_wmma_f32_16x16x32_bf16(false, a, false, b,
                                                 (short)0, c, false, false);
}

__device__ __forceinline__ v8f v8f_zero() {
  v8f z = {0.f, 0.f, 0.f, 0.f, 0.f, 0.f, 0.f, 0.f};
  return z;
}

// ---------------------------------------------------------------------------
// async global->LDS copy of one 16B chunk (per lane), with sync fallback
// ---------------------------------------------------------------------------
__device__ __forceinline__ void cp16(const bf16_t* __restrict__ g,
                                     bf16_t* __restrict__ l) {
#if HAVE_ASYNC_LDS
  __builtin_amdgcn_global_load_async_to_lds_b128((as1_v4i*)g, (as3_v4i*)l, 0, 0);
#else
  *(v8bf*)l = *(const v8bf*)g;
#endif
}

__device__ __forceinline__ void wait_async_le8() {
#if HAVE_ASYNC_LDS
#if __has_builtin(__builtin_amdgcn_s_wait_asynccnt)
  __builtin_amdgcn_s_wait_asynccnt(8);
#else
  asm volatile("s_wait_asynccnt 0x8" ::: "memory");
#endif
#endif
}

__device__ __forceinline__ void wait_async_le0() {
#if HAVE_ASYNC_LDS
#if __has_builtin(__builtin_amdgcn_s_wait_asynccnt)
  __builtin_amdgcn_s_wait_asynccnt(0);
#else
  asm volatile("s_wait_asynccnt 0x0" ::: "memory");
#endif
#endif
}

// ---------------------------------------------------------------------------
// Prep kernels
// ---------------------------------------------------------------------------
__global__ void k_cast_x(const float* __restrict__ x, bf16_t* __restrict__ xb, int n) {
  int i = blockIdx.x * 256 + threadIdx.x;
  if (i < n) xb[i] = (bf16_t)x[i];
}

// Wcat1 rows: [0,1024): qkv_w q-rows * HD^-0.5 ; [1024,3072): k,v rows;
// [3072,3328): ql_w * HDL^-0.5 ; [3328,3584): kl_w ; [3584,3840): vl_w
__global__ void k_build_w1(const float* __restrict__ qkv_w, const float* __restrict__ ql_w,
                           const float* __restrict__ kl_w,  const float* __restrict__ vl_w,
                           bf16_t* __restrict__ w1) {
  int idx = blockIdx.x * 256 + threadIdx.x;
  if (idx >= N1 * K1) return;
  int row = idx / K1, col = idx - row * K1;
  float v;
  if (row < 3 * DIM) {
    v = qkv_w[idx];
    if (row < DIM) v *= 0.17677669529663687f;               // 1/sqrt(32)
  } else if (row < 3 * DIM + LITE) {
    v = ql_w[(row - 3 * DIM) * K1 + col] * 0.35355339059327373f; // 1/sqrt(8)
  } else if (row < 3 * DIM + 2 * LITE) {
    v = kl_w[(row - (3 * DIM + LITE)) * K1 + col];
  } else {
    v = vl_w[(row - (3 * DIM + 2 * LITE)) * K1 + col];
  }
  w1[idx] = (bf16_t)v;
}

__global__ void k_build_b1(const float* __restrict__ qkv_b, float* __restrict__ b1) {
  int i = blockIdx.x * 256 + threadIdx.x;
  if (i >= N1) return;
  float v = 0.f;
  if (i < 3 * DIM) { v = qkv_b[i]; if (i < DIM) v *= 0.17677669529663687f; }
  b1[i] = v;
}

// Wcat2[n][k]: k<1024 -> proj_w[n][k]; else gamma * (proj_w @ diffproj_w)[n][k-1024]
__global__ void k_build_w2(const float* __restrict__ proj_w, const float* __restrict__ dpw,
                           const float* __restrict__ gp, bf16_t* __restrict__ w2) {
  int idx = blockIdx.x * 256 + threadIdx.x;
  if (idx >= N2 * K2) return;
  int n = idx / K2, k = idx - n * K2;
  float v;
  if (k < DIM) {
    v = proj_w[n * DIM + k];
  } else {
    int kk = k - DIM;
    float acc = 0.f;
    for (int j = 0; j < DIM; ++j) acc = fmaf(proj_w[n * DIM + j], dpw[j * LITE + kk], acc);
    v = gp[0] * acc;
  }
  w2[idx] = (bf16_t)v;
}

// bias2[n] = proj_b[n] + gamma * sum_j proj_w[n][j] * diffproj_b[j]
__global__ void k_build_b2(const float* __restrict__ proj_w, const float* __restrict__ proj_b,
                           const float* __restrict__ dpb, const float* __restrict__ gp,
                           float* __restrict__ b2) {
  int n = blockIdx.x * 256 + threadIdx.x;
  if (n >= N2) return;
  float acc = 0.f;
  for (int j = 0; j < DIM; ++j) acc = fmaf(proj_w[n * DIM + j], dpb[j], acc);
  b2[n] = proj_b[n] + gp[0] * acc;
}

// btab[h][i][j] = rpb_table[rel_pos_index[i*49+j]][h]
__global__ void k_build_btab(const float* __restrict__ rpb, const int* __restrict__ rpi,
                             float* __restrict__ btab) {
  int idx = blockIdx.x * 256 + threadIdx.x;
  if (idx >= NHEAD * NW * NW) return;
  int h = idx / (NW * NW), ij = idx - h * (NW * NW);
  btab[idx] = rpb[rpi[ij] * NHEAD + h];
}

// ---------------------------------------------------------------------------
// WMMA NT-GEMM: C[M,N] = A[M,K] * W[N,K]^T + bias[N]
// 256 threads = 8 waves (2 M x 4 N); block tile 128x128; wave 64x32.
// K-slab 64, double-buffered LDS fed by async global->LDS b128 copies
// (8 per thread per slab => s_wait_asynccnt <= 8 pipelines the next slab).
// Dynamic LDS: 2 * (128*72 + 128*72) * 2B = 72 KB.
// ---------------------------------------------------------------------------
template <bool OUT_F32>
__global__ __launch_bounds__(256) void gemm_nt(const bf16_t* __restrict__ A,
                                               const bf16_t* __restrict__ W,
                                               const float* __restrict__ bias,
                                               void* __restrict__ Cout,
                                               int M, int N, int K) {
  extern __shared__ __align__(16) bf16_t sm[];
  bf16_t* buf[2];
  buf[0] = sm;                       // [A(128x72) | B(128x72)]
  buf[1] = sm + 2 * BM * LDT;

  const int tid = threadIdx.x;
  const int lane = tid & 31;
  const int wave = tid >> 5;
  const int wm = wave & 1;           // 2 waves along M
  const int wn = wave >> 1;          // 4 waves along N
  const int m0 = blockIdx.y * BM;
  const int n0 = blockIdx.x * BN;

  const bf16_t* gA = A + (size_t)m0 * K;
  const bf16_t* gB = W + (size_t)n0 * K;

  v8f acc[4][2];
#pragma unroll
  for (int i = 0; i < 4; ++i)
#pragma unroll
    for (int j = 0; j < 2; ++j) acc[i][j] = v8f_zero();

  // stage one K-slab into LDS buffer p (8 async b128 per thread)
  auto stage = [&](int kt, int p) {
    const bf16_t* ga = gA + kt * BK;
    const bf16_t* gb = gB + kt * BK;
    bf16_t* la = buf[p];
    bf16_t* lb = buf[p] + BM * LDT;
#pragma unroll
    for (int i = 0; i < 4; ++i) {
      const int chunk = tid + i * 256;        // 0..1023
      const int row = chunk >> 3;             // 8 chunks (16B) per 64-elem row
      const int c8  = (chunk & 7) * 8;
      cp16(ga + (size_t)row * K + c8, la + row * LDT + c8);
    }
#pragma unroll
    for (int i = 0; i < 4; ++i) {
      const int chunk = tid + i * 256;
      const int row = chunk >> 3;
      const int c8  = (chunk & 7) * 8;
      cp16(gb + (size_t)row * K + c8, lb + row * LDT + c8);
    }
  };

  const int nk = K / BK;
  stage(0, 0);
  for (int kt = 0; kt < nk; ++kt) {
    if (kt + 1 < nk) {
      stage(kt + 1, (kt + 1) & 1);
      wait_async_le8();   // slab kt resident; slab kt+1 still in flight
    } else {
      wait_async_le0();
    }
    __syncthreads();

    const bf16_t* wA = buf[kt & 1] + (size_t)(wm * 64) * LDT;
    const bf16_t* wB = buf[kt & 1] + (size_t)BM * LDT + (size_t)(wn * 32) * LDT;
#pragma unroll
    for (int ks = 0; ks < BK; ks += 32) {
      v16bf a[4], b[2];
#pragma unroll
      for (int i = 0; i < 4; ++i) a[i] = frag_load(wA + (size_t)(16 * i) * LDT + ks, LDT, lane);
#pragma unroll
      for (int j = 0; j < 2; ++j) b[j] = frag_load(wB + (size_t)(16 * j) * LDT + ks, LDT, lane);
#pragma unroll
      for (int i = 0; i < 4; ++i)
#pragma unroll
        for (int j = 0; j < 2; ++j) acc[i][j] = wmma_bf16(a[i], b[j], acc[i][j]);
    }
    __syncthreads();   // protect buffer reuse before next stage overwrites
  }

  const int mw = m0 + wm * 64;
  const int nwc = n0 + wn * 32;
  const int rbase = (lane >> 4) * 8;
  const int cl = lane & 15;
#pragma unroll
  for (int i = 0; i < 4; ++i)
#pragma unroll
    for (int j = 0; j < 2; ++j) {
      const int col = nwc + 16 * j + cl;
      const float bv = bias[col];
#pragma unroll
      for (int r = 0; r < 8; ++r) {
        const int row = mw + 16 * i + rbase + r;
        const float v = acc[i][j][r] + bv;
        if (OUT_F32) ((float*)Cout)[(size_t)row * N + col] = v;
        else         ((bf16_t*)Cout)[(size_t)row * N + col] = (bf16_t)v;
      }
    }
}

// ---------------------------------------------------------------------------
// Attention kernel: one block per (window-batch b, head h). 128 thr = 4 waves.
// Scores via WMMA (49 padded to 64), softmax per row, P/V & diff/Vl via WMMA.
// Static LDS = 54 KB.
// ---------------------------------------------------------------------------
__global__ __launch_bounds__(128) void attn_kernel(const bf16_t* __restrict__ Y,
                                                   const float* __restrict__ mask,
                                                   const float* __restrict__ btab,
                                                   const float* __restrict__ lam1p,
                                                   const float* __restrict__ lam2p,
                                                   bf16_t* __restrict__ A2) {
  __shared__ __align__(16) bf16_t sQ[64][32];    // scale pre-folded
  __shared__ __align__(16) bf16_t sK[64][32];
  __shared__ __align__(16) bf16_t sVT[32][64];   // V^T (k-padded w/ zeros)
  __shared__ __align__(16) bf16_t sQl[64][32];   // K padded 8->32 w/ zeros
  __shared__ __align__(16) bf16_t sKl[64][32];
  __shared__ __align__(16) bf16_t sVlT[16][64];  // Vl^T (N and K padded)
  __shared__ __align__(16) float  sS[64][64];    // score scratch (reused)
  __shared__ __align__(16) bf16_t sP[64][64];    // softmax(main)
  __shared__ __align__(16) bf16_t sPl[64][64];   // softmax(lite)

  const int b = blockIdx.x;             // 0..511
  const int h = blockIdx.y;             // 0..31
  const int tid = threadIdx.x;
  const int lane = tid & 31;
  const int wv = tid >> 5;              // wave id -> rows 16wv..16wv+15
  const int win = b & (NUMWIN - 1);     // b % 64 per reference reshape
  const bf16_t* Yb = Y + (size_t)b * NW * N1;
  const bf16_t z = (bf16_t)0.f;

  for (int idx = tid; idx < 64 * 32; idx += 128) {
    const int r = idx >> 5, c = idx & 31;
    bf16_t q0 = z, k0 = z, v0 = z, ql0 = z, kl0 = z;
    if (r < NW) {
      const bf16_t* row = Yb + (size_t)r * N1;
      q0 = row[h * 32 + c];
      k0 = row[DIM + h * 32 + c];
      v0 = row[2 * DIM + h * 32 + c];
      if (c < 8) {
        ql0 = row[3 * DIM + h * 8 + c];
        kl0 = row[3 * DIM + LITE + h * 8 + c];
      }
    }
    sQ[r][c] = q0; sK[r][c] = k0; sVT[c][r] = v0; sQl[r][c] = ql0; sKl[r][c] = kl0;
  }
  for (int idx = tid; idx < 16 * 64; idx += 128) {
    const int nn = idx >> 6, r = idx & 63;
    bf16_t v = z;
    if (nn < 8 && r < NW) v = Yb[(size_t)r * N1 + 3 * DIM + 2 * LITE + h * 8 + nn];
    sVlT[nn][r] = v;
  }
  __syncthreads();

  const float l1 = 1.f / (1.f + __expf(-lam1p[0]));
  const float l2 = 1.f / (1.f + __expf(-lam2p[0]));
  const float* mrowb = mask + (size_t)win * NW * NW;
  const float* browb = btab + (size_t)h * NW * NW;

  // ---- main branch: scores -> softmax -> sP ----
  {
    const v16bf aq = frag_load(&sQ[wv * 16][0], 32, lane);
    const int rr = wv * 16 + ((lane >> 4) * 8);
    const int cb = lane & 15;
#pragma unroll
    for (int nt = 0; nt < 4; ++nt) {
      v8f c = v8f_zero();
      c = wmma_bf16(aq, frag_load(&sK[nt * 16][0], 32, lane), c);
      const int cc = nt * 16 + cb;
#pragma unroll
      for (int r = 0; r < 8; ++r) sS[rr + r][cc] = c[r];
    }
  }
  __syncthreads();
  if (tid < NW) {
    const int i = tid;
    const float* mrow = mrowb + (size_t)i * NW;
    const float* brow = browb + (size_t)i * NW;
    float mx = -1e30f;
    for (int j = 0; j < NW; ++j) {
      const float s = sS[i][j] + brow[j] + mrow[j];
      sS[i][j] = s;
      mx = fmaxf(mx, s);
    }
    float sum = 0.f;
    for (int j = 0; j < NW; ++j) sum += __expf(sS[i][j] - mx);
    const float inv = 1.f / sum;
    for (int j = 0; j < 64; ++j)
      sP[i][j] = (j < NW) ? (bf16_t)(__expf(sS[i][j] - mx) * inv) : z;
  } else if (tid < 64) {
    for (int j = 0; j < 64; ++j) sP[tid][j] = z;
  }
  __syncthreads();

  // ---- lite branch: scores -> softmax -> sPl ----
  {
    const v16bf aql = frag_load(&sQl[wv * 16][0], 32, lane);
    const int rr = wv * 16 + ((lane >> 4) * 8);
    const int cb = lane & 15;
#pragma unroll
    for (int nt = 0; nt < 4; ++nt) {
      v8f c = v8f_zero();
      c = wmma_bf16(aql, frag_load(&sKl[nt * 16][0], 32, lane), c);
      const int cc = nt * 16 + cb;
#pragma unroll
      for (int r = 0; r < 8; ++r) sS[rr + r][cc] = c[r];
    }
  }
  __syncthreads();
  if (tid < NW) {
    const int i = tid;
    const float* mrow = mrowb + (size_t)i * NW;
    const float* brow = browb + (size_t)i * NW;
    float mx = -1e30f;
    for (int j = 0; j < NW; ++j) {
      const float s = sS[i][j] + brow[j] + mrow[j];
      sS[i][j] = s;
      mx = fmaxf(mx, s);
    }
    float sum = 0.f;
    for (int j = 0; j < NW; ++j) sum += __expf(sS[i][j] - mx);
    const float inv = 1.f / sum;
    for (int j = 0; j < 64; ++j)
      sPl[i][j] = (j < NW) ? (bf16_t)(__expf(sS[i][j] - mx) * inv) : z;
  } else if (tid < 64) {
    for (int j = 0; j < 64; ++j) sPl[tid][j] = z;
  }
  __syncthreads();

  // ---- outputs: x_orig = P@V ; dv = (l1*Pl - l2*P)@Vl ----
  v8f co0 = v8f_zero(), co1 = v8f_zero(), cd = v8f_zero();
#pragma unroll
  for (int ks = 0; ks < 2; ++ks) {
    const v16bf ap  = frag_load(&sP[wv * 16][ks * 32], 64, lane);
    const v16bf apl = frag_load(&sPl[wv * 16][ks * 32], 64, lane);
    v16bf ad;
#pragma unroll
    for (int e = 0; e < 16; ++e)
      ad[e] = (bf16_t)(l1 * (float)apl[e] - l2 * (float)ap[e]);
    co0 = wmma_bf16(ap, frag_load(&sVT[0][ks * 32],  64, lane), co0);
    co1 = wmma_bf16(ap, frag_load(&sVT[16][ks * 32], 64, lane), co1);
    cd  = wmma_bf16(ad, frag_load(&sVlT[0][ks * 32], 64, lane), cd);
  }

  bf16_t* Ab = A2 + (size_t)b * NW * K2;
  const int rr = wv * 16 + ((lane >> 4) * 8);
  const int cc = lane & 15;
#pragma unroll
  for (int r = 0; r < 8; ++r) {
    const int row = rr + r;
    if (row < NW) {
      bf16_t* orow = Ab + (size_t)row * K2;
      orow[h * 32 + cc]      = (bf16_t)co0[r];
      orow[h * 32 + 16 + cc] = (bf16_t)co1[r];
      if (cc < 8) orow[DIM + h * 8 + cc] = (bf16_t)cd[r];
    }
  }
}

// ---------------------------------------------------------------------------
// Host-side orchestration
// ---------------------------------------------------------------------------
extern "C" void kernel_launch(void* const* d_in, const int* in_sizes, int n_in,
                              void* d_out, int out_size, void* d_ws, size_t ws_size,
                              hipStream_t stream) {
  (void)in_sizes; (void)n_in; (void)out_size; (void)ws_size;
  const float* x      = (const float*)d_in[0];
  const float* mask   = (const float*)d_in[1];
  const float* qkv_w  = (const float*)d_in[2];
  const float* qkv_b  = (const float*)d_in[3];
  const float* proj_w = (const float*)d_in[4];
  const float* proj_b = (const float*)d_in[5];
  const float* rpb    = (const float*)d_in[6];
  const float* ql_w   = (const float*)d_in[7];
  const float* kl_w   = (const float*)d_in[8];
  const float* vl_w   = (const float*)d_in[9];
  const float* lam1   = (const float*)d_in[10];
  const float* lam2   = (const float*)d_in[11];
  const float* dpw    = (const float*)d_in[12];
  const float* dpb    = (const float*)d_in[13];
  const float* gamma  = (const float*)d_in[14];
  const int*   rpi    = (const int*)d_in[15];

  char* ws = (char*)d_ws;
  size_t off = 0;
  auto carve = [&](size_t bytes) -> void* {
    void* p = (void*)(ws + off);
    off += (bytes + 255) & ~(size_t)255;
    return p;
  };
  bf16_t* xbf  = (bf16_t*)carve((size_t)MTOK * K1 * sizeof(bf16_t));   // 51.4 MB
  bf16_t* w1   = (bf16_t*)carve((size_t)N1 * K1 * sizeof(bf16_t));     // 7.9 MB
  float*  b1   = (float*) carve((size_t)N1 * sizeof(float));
  bf16_t* y1   = (bf16_t*)carve((size_t)MTOK * N1 * sizeof(bf16_t));   // 192.7 MB
  bf16_t* a2   = (bf16_t*)carve((size_t)MTOK * K2 * sizeof(bf16_t));   // 64.2 MB
  bf16_t* w2   = (bf16_t*)carve((size_t)N2 * K2 * sizeof(bf16_t));     // 2.6 MB
  float*  b2   = (float*) carve((size_t)N2 * sizeof(float));
  float*  btab = (float*) carve((size_t)NHEAD * NW * NW * sizeof(float));

  // prep
  {
    int n = MTOK * K1;
    k_cast_x<<<(n + 255) / 256, 256, 0, stream>>>(x, xbf, n);
  }
  k_build_w1<<<(N1 * K1 + 255) / 256, 256, 0, stream>>>(qkv_w, ql_w, kl_w, vl_w, w1);
  k_build_b1<<<(N1 + 255) / 256, 256, 0, stream>>>(qkv_b, b1);
  k_build_w2<<<(N2 * K2 + 255) / 256, 256, 0, stream>>>(proj_w, dpw, gamma, w2);
  k_build_b2<<<(N2 + 255) / 256, 256, 0, stream>>>(proj_w, proj_b, dpb, gamma, b2);
  k_build_btab<<<(NHEAD * NW * NW + 255) / 256, 256, 0, stream>>>(rpb, rpi, btab);

  const size_t gemm_lds = (size_t)4 * BM * LDT * sizeof(bf16_t);  // 73728 B

  // fused qkv + lite projection: Y[25088,3840]
  gemm_nt<false><<<dim3(N1 / BN, MTOK / BM), 256, gemm_lds, stream>>>(
      xbf, w1, b1, (void*)y1, MTOK, N1, K1);

  // differential window attention -> concat activation A2[25088,1280]
  attn_kernel<<<dim3(BATCH, NHEAD), 128, 0, stream>>>(y1, mask, btab, lam1, lam2, a2);

  // folded output projection -> d_out (fp32)
  gemm_nt<true><<<dim3(N2 / BN, MTOK / BM), 256, gemm_lds, stream>>>(
      a2, w2, b2, d_out, MTOK, N2, K2);
}
